// RelationalTransformer_77962246357427
// MI455X (gfx1250) — compile-verified, matrix operand order checked by reading, not run on hardware
//
#include <hip/hip_runtime.h>
#include <stdint.h>

// ---------------------------------------------------------------------------
// MI455X / gfx1250 relational transformer.
// - all contractions on v_wmma_f32_16x16x32_bf16 (wave32 matrix pipe)
// - GEMM weight tiles staged to LDS with global_load_async_to_lds_b128
//   (ASYNCcnt-tracked async copy, ping-pong double buffered, peeled tail)
// - raw s_barrier_signal/wait (no full fence) so prefetched global loads
//   stay in flight across barriers
// - B fragments software-pipelined (ds_loads for frag j+1 issued before
//   WMMAs of frag j) to avoid full dscnt drains
// - software-pipelined attention (V early, K prefetch under softmax VALU)
// ---------------------------------------------------------------------------

typedef __bf16 bf16;
typedef __attribute__((ext_vector_type(16))) __bf16 v16bf;
typedef __attribute__((ext_vector_type(8)))  float  v8f;

#define BATCH  4
#define SEQ    1024
#define DIM    512
#define NHEAD  8
#define HDIM   64
#define DFFN   2048
#define NLAYER 4

// float -> bf16, round-to-nearest-even (pure bit math)
__device__ inline bf16 f2bf(float x) {
  uint32_t u = __builtin_bit_cast(uint32_t, x);
  uint32_t r = u + 0x7fffu + ((u >> 16) & 1u);
  uint16_t s = (uint16_t)(r >> 16);
  return __builtin_bit_cast(bf16, s);
}
__device__ inline float bf2f(bf16 h) {
  uint16_t s = __builtin_bit_cast(uint16_t, h);
  uint32_t u = ((uint32_t)s) << 16;
  return __builtin_bit_cast(float, u);
}

union Frag16 { v16bf v; uint4 q[2]; };
union Pack8  { uint4 q; bf16 h[8]; };

// CDNA5 16-bit A/B fragment: lane l holds row/col (l&15); its 16 elements are
// K = base..base+7 and base+16..base+23 with base = 8*(l>=16). Caller passes
// p = row_start + k_chunk + base  -> two contiguous 16B loads.
__device__ inline v16bf ld_frag(const bf16* p) {
  Frag16 f;
  f.q[0] = *reinterpret_cast<const uint4*>(p);
  f.q[1] = *reinterpret_cast<const uint4*>(p + 16);
  return f.v;
}

__device__ inline v8f wmma_bf16(v16bf a, v16bf b, v8f c) {
  // emits v_wmma_f32_16x16x32_bf16
  return __builtin_amdgcn_wmma_f32_16x16x32_bf16(false, a, false, b,
                                                 (short)0, c, false, false);
}

// async copy of 16B per lane: global -> LDS, tracked by ASYNCcnt
__device__ inline void async_copy_b128(uint32_t lds_addr, const void* gaddr) {
  asm volatile("global_load_async_to_lds_b128 %0, %1, off"
               :: "v"(lds_addr), "v"(gaddr) : "memory");
}
__device__ inline void wait_async_le1() {
  asm volatile("s_wait_asynccnt 0x1" ::: "memory");
}
__device__ inline void wait_async_le0() {
  asm volatile("s_wait_asynccnt 0x0" ::: "memory");
}
// execution-only workgroup barrier (no loadcnt fence: global prefetches may
// stay outstanding). Acquire flavor: async data already waited per-wave.
__device__ inline void wg_barrier_acq() {
  asm volatile("s_barrier_signal -1\n\t"
               "s_barrier_wait -1" ::: "memory");
}
// Release flavor: drain own LDS reads before allowing DMA to overwrite.
__device__ inline void wg_barrier_rel() {
  asm volatile("s_wait_dscnt 0x0\n\t"
               "s_barrier_signal -1\n\t"
               "s_barrier_wait -1" ::: "memory");
}

// ---------------------------------------------------------------------------
// Weight convert: W (K,N) f32 row-major -> Wt (N,K) bf16 (W^T), tiled 32x32
// via LDS so both global sides are coalesced. Grid = (K/32)*(N/32) blocks.
// ---------------------------------------------------------------------------
__global__ __launch_bounds__(256)
void convw_kernel(const float* __restrict__ W, bf16* __restrict__ Wt,
                  int K, int N)
{
  __shared__ float tile[32][33];
  int nt = N >> 5;
  int tk = blockIdx.x / nt;
  int tn = blockIdx.x - tk * nt;
  int c = threadIdx.x & 31;
  int r8 = threadIdx.x >> 5;            // 0..7
#pragma unroll
  for (int it = 0; it < 4; ++it) {
    int r = r8 + it * 8;
    tile[r][c] = W[(size_t)(tk * 32 + r) * N + tn * 32 + c];
  }
  __syncthreads();
#pragma unroll
  for (int it = 0; it < 4; ++it) {
    int r = r8 + it * 8;
    Wt[(size_t)(tn * 32 + r) * K + tk * 32 + c] = f2bf(tile[c][r]);
  }
}

// ---------------------------------------------------------------------------
// RMSNorm: one token per 256-thread block, D = 512, bf16 output
// ---------------------------------------------------------------------------
__global__ __launch_bounds__(256)
void rmsnorm_kernel(const float* __restrict__ X, const float* __restrict__ W,
                    bf16* __restrict__ Y)
{
  __shared__ float red[8];
  int row = blockIdx.x;
  int t = threadIdx.x;
  const float* x = X + (size_t)row * DIM;
  float a = x[t], b = x[t + 256];
  float ss = a * a + b * b;
#pragma unroll
  for (int off = 16; off > 0; off >>= 1) ss += __shfl_xor(ss, off, 32);
  if ((t & 31) == 0) red[t >> 5] = ss;
  __syncthreads();
  float tot = 0.f;
#pragma unroll
  for (int i = 0; i < 8; ++i) tot += red[i];
  float inv = rsqrtf(tot * (1.0f / DIM) + 1.1920929e-07f);
  bf16* y = Y + (size_t)row * DIM;
  y[t]       = f2bf(a * inv * W[t]);
  y[t + 256] = f2bf(b * inv * W[t + 256]);
}

// ---------------------------------------------------------------------------
// bf16 GEMM: C(M,N) = A(M,K) * Bt(N,K)^T, f32 accumulate.
// Block = 8 waves sharing one 64-wide N tile; wave computes 32(M) x 64(N).
// B chunk (64x32 bf16 = 4KB) staged to LDS via global_load_async_to_lds_b128,
// ping-pong buffered, branch-free steady state + peeled tail.
// mode 0: bf16 row-major (ldc) | mode 1: bf16 into Vt (B,DIM,SEQ) | mode 2: f32 +=
// ---------------------------------------------------------------------------
__global__ __launch_bounds__(256)
void gemm_kernel(const bf16* __restrict__ A, const bf16* __restrict__ Bt,
                 int M, int N, int K, void* __restrict__ Cout,
                 int mode, int ldc)
{
  __shared__ __align__(16) bf16 bstage[2][64 * 32];   // 2 x 4KB ping-pong

  int tid = threadIdx.x;
  int lane = tid & 31;
  int wv = tid >> 5;                       // wave 0..7
  int nb = N >> 6;
  int bm = blockIdx.x / nb;
  int nt = blockIdx.x - bm * nb;
  int mt = bm * 8 + wv;                    // 32-row tile index
  int ll = lane & 15;
  int hi = lane >> 4;
  int boff = hi << 3;
  int n0 = nt * 64;
  int m0 = mt * 32;

  // async-copy mapping: 256 threads x 16B cover the 64x32 bf16 chunk
  int crow = tid >> 2;                     // 0..63  (n-local row)
  int cseg = (tid & 3) << 3;               // element offset 0/8/16/24
  const bf16* gsrc = Bt + (size_t)(n0 + crow) * K + cseg;
  uint32_t ldst0 = (uint32_t)(uintptr_t)(&bstage[0][crow * 32 + cseg]);
  uint32_t ldst1 = (uint32_t)(uintptr_t)(&bstage[1][crow * 32 + cseg]);

  async_copy_b128(ldst0, gsrc);            // chunk 0 in flight

  const bf16* arow0 = A + (size_t)(m0 + ll) * K;
  const bf16* arow1 = A + (size_t)(m0 + 16 + ll) * K;

  v8f acc[2][4] = {};
  v16bf a0 = ld_frag(arow0 + boff);
  v16bf a1 = ld_frag(arow1 + boff);

  // B fragments software-pipelined: frag j+1's ds_loads issue before the
  // WMMAs consuming frag j, so dscnt waits are partial, not full drains.
  auto compute = [&](int cur, v16bf ca0, v16bf ca1) {
    const bf16* bb = &bstage[cur][0] + ll * 32 + boff;
    v16bf bf0 = ld_frag(bb + 0 * 16 * 32);
    v16bf bf1 = ld_frag(bb + 1 * 16 * 32);
    acc[0][0] = wmma_bf16(ca0, bf0, acc[0][0]);
    acc[1][0] = wmma_bf16(ca1, bf0, acc[1][0]);
    v16bf bf2 = ld_frag(bb + 2 * 16 * 32);
    acc[0][1] = wmma_bf16(ca0, bf1, acc[0][1]);
    acc[1][1] = wmma_bf16(ca1, bf1, acc[1][1]);
    v16bf bf3 = ld_frag(bb + 3 * 16 * 32);
    acc[0][2] = wmma_bf16(ca0, bf2, acc[0][2]);
    acc[1][2] = wmma_bf16(ca1, bf2, acc[1][2]);
    acc[0][3] = wmma_bf16(ca0, bf3, acc[0][3]);
    acc[1][3] = wmma_bf16(ca1, bf3, acc[1][3]);
  };

  int nk = K >> 5;
  for (int i = 0; i < nk - 1; ++i) {       // branch-free steady state
    int cur = i & 1;
    async_copy_b128(cur ? ldst0 : ldst1, gsrc + (size_t)(i + 1) * 32);
    wait_async_le1();                      // chunk i resident (in-order)
    wg_barrier_acq();                      // all waves' DMA arrived
    v16bf a0n = ld_frag(arow0 + (i + 1) * 32 + boff);   // stays in flight
    v16bf a1n = ld_frag(arow1 + (i + 1) * 32 + boff);
    compute(cur, a0, a1);
    wg_barrier_rel();                      // reads drained before overwrite
    a0 = a0n; a1 = a1n;
  }
  wait_async_le0();                        // tail: last chunk resident
  wg_barrier_acq();
  compute((nk - 1) & 1, a0, a1);

  if (mode == 0) {
    bf16* C = (bf16*)Cout;
#pragma unroll
    for (int t = 0; t < 2; ++t)
#pragma unroll
      for (int j = 0; j < 4; ++j)
#pragma unroll
        for (int v = 0; v < 8; ++v) {
          int row = m0 + t * 16 + v + 8 * hi;
          int col = n0 + j * 16 + ll;
          C[(size_t)row * ldc + col] = f2bf(acc[t][j][v]);
        }
  } else if (mode == 1) {
    bf16* C = (bf16*)Cout;                 // Vt laid out (B, DIM, SEQ)
#pragma unroll
    for (int t = 0; t < 2; ++t)
#pragma unroll
      for (int j = 0; j < 4; ++j)
#pragma unroll
        for (int v = 0; v < 8; ++v) {
          int row = m0 + t * 16 + v + 8 * hi;   // token index b*SEQ+s
          int col = n0 + j * 16 + ll;           // feature index
          int b = row >> 10;
          int s = row & (SEQ - 1);
          C[((size_t)b * DIM + col) * SEQ + s] = f2bf(acc[t][j][v]);
        }
  } else {
    float* C = (float*)Cout;
#pragma unroll
    for (int t = 0; t < 2; ++t)
#pragma unroll
      for (int j = 0; j < 4; ++j)
#pragma unroll
        for (int v = 0; v < 8; ++v) {
          int row = m0 + t * 16 + v + 8 * hi;
          int col = n0 + j * 16 + ll;
          size_t idx = (size_t)row * ldc + col;
          C[idx] = C[idx] + acc[t][j][v];
        }
  }
}

// ---------------------------------------------------------------------------
// Masked flash attention with TRANSPOSED scores.
//   S^T = K_tile * Q^T  (A = K rows, B = Q rows; identical fragment loads)
//   O^T = V^T  * P^T    (score C-layout == P^T B-layout over a 32-key chunk)
// One wave per 16-query tile of one (b,h); lane = query column. V fragments
// load before the score WMMAs; next chunk's K fragments prefetch under the
// softmax VALU section.
// ---------------------------------------------------------------------------
__global__ __launch_bounds__(256)
void attn_kernel(const bf16* __restrict__ Q, const bf16* __restrict__ Km,
                 const bf16* __restrict__ Vt, const unsigned char* __restrict__ M,
                 bf16* __restrict__ O)
{
  const float scale = 0.125f;               // HD^-0.5, HD=64
  int lane = threadIdx.x & 31;
  int wave = blockIdx.x * (blockDim.x >> 5) + (threadIdx.x >> 5);
  int qt = wave & (SEQ / 16 - 1);           // 64 q-tiles
  int h  = (wave >> 6) & (NHEAD - 1);
  int b  = wave >> 9;
  int ll = lane & 15;
  int hi = lane >> 4;
  int boff = hi << 3;
  int q0 = qt * 16;

  const bf16* qrow = Q + ((size_t)b * SEQ + q0 + ll) * DIM + h * HDIM;
  v16bf qa = ld_frag(qrow + boff);
  v16bf qb = ld_frag(qrow + 32 + boff);

  const unsigned char* mrow = M + ((size_t)b * SEQ + q0 + ll) * SEQ;
  const bf16* kbase = Km + ((size_t)b * SEQ + ll) * DIM + h * HDIM;
  const bf16* vbase = Vt + ((size_t)b * DIM + h * HDIM + ll) * SEQ + boff;

  v8f acc[4] = {};                          // O^T accumulators, d-chunks of 16
  float rmax = -3.0e38f;
  float rsum = 0.0f;

  // preload K fragments for chunk 0 (two 16-key tiles x two 32-hd chunks)
  v16bf kc0a = ld_frag(kbase + boff);
  v16bf kc0b = ld_frag(kbase + 32 + boff);
  v16bf kc1a = ld_frag(kbase + (size_t)16 * DIM + boff);
  v16bf kc1b = ld_frag(kbase + (size_t)16 * DIM + 32 + boff);

  for (int kc = 0; kc < SEQ; kc += 32) {
    // V fragments for this chunk: independent of scores, load early
    v16bf va0 = ld_frag(vbase + (size_t)(0 * 16) * SEQ + kc);
    v16bf va1 = ld_frag(vbase + (size_t)(1 * 16) * SEQ + kc);
    v16bf va2 = ld_frag(vbase + (size_t)(2 * 16) * SEQ + kc);
    v16bf va3 = ld_frag(vbase + (size_t)(3 * 16) * SEQ + kc);

    unsigned long long mb0 =
        *reinterpret_cast<const unsigned long long*>(mrow + kc + boff);
    unsigned long long mb1 =
        *reinterpret_cast<const unsigned long long*>(mrow + kc + 16 + boff);

    v8f s0 = {}, s1 = {};
    s0 = wmma_bf16(kc0a, qa, s0);
    s0 = wmma_bf16(kc0b, qb, s0);
    s1 = wmma_bf16(kc1a, qa, s1);
    s1 = wmma_bf16(kc1b, qb, s1);

    // prefetch next chunk's K fragments under the softmax VALU section
    if (kc + 32 < SEQ) {
      const bf16* knext = kbase + (size_t)(kc + 32) * DIM;
      kc0a = ld_frag(knext + boff);
      kc0b = ld_frag(knext + 32 + boff);
      kc1a = ld_frag(knext + (size_t)16 * DIM + boff);
      kc1b = ld_frag(knext + (size_t)16 * DIM + 32 + boff);
    }

    float f0[8], f1[8];
    float mx = rmax;
#pragma unroll
    for (int v = 0; v < 8; ++v) {
      bool m0 = (mb0 >> (8 * v)) & 1ull;
      bool m1 = (mb1 >> (8 * v)) & 1ull;
      f0[v] = m0 ? s0[v] * scale : -1.0e30f;
      f1[v] = m1 ? s1[v] * scale : -1.0e30f;
      mx = fmaxf(mx, fmaxf(f0[v], f1[v]));
    }
    mx = fmaxf(mx, __shfl_xor(mx, 16, 32)); // both half-wave lanes of a query
    float r = __expf(rmax - mx);
    rmax = mx;

    float lsum = 0.0f;
    v16bf pf;                               // P^T B-fragment for this chunk
#pragma unroll
    for (int v = 0; v < 8; ++v) {
      bool m0 = (mb0 >> (8 * v)) & 1ull;
      bool m1 = (mb1 >> (8 * v)) & 1ull;
      float e0 = m0 ? __expf(f0[v] - mx) : 0.0f;
      float e1 = m1 ? __expf(f1[v] - mx) : 0.0f;
      lsum += e0 + e1;
      pf[v]     = f2bf(e0);                 // keys boff..boff+7
      pf[v + 8] = f2bf(e1);                 // keys boff+16..boff+23
    }
    lsum += __shfl_xor(lsum, 16, 32);
    rsum = rsum * r + lsum;

#pragma unroll
    for (int j = 0; j < 4; ++j)
#pragma unroll
      for (int v = 0; v < 8; ++v) acc[j][v] *= r;
    acc[0] = wmma_bf16(va0, pf, acc[0]);
    acc[1] = wmma_bf16(va1, pf, acc[1]);
    acc[2] = wmma_bf16(va2, pf, acc[2]);
    acc[3] = wmma_bf16(va3, pf, acc[3]);
  }

  float inv = 1.0f / fmaxf(rsum, 1.0e-20f);
  bf16* orow = O + ((size_t)b * SEQ + q0 + ll) * DIM + h * HDIM;
#pragma unroll
  for (int j = 0; j < 4; ++j) {
    Pack8 p;
#pragma unroll
    for (int v = 0; v < 8; ++v) p.h[v] = f2bf(acc[j][v] * inv);
    *reinterpret_cast<uint4*>(orow + j * 16 + boff) = p.q;   // 16B store
  }
}

// ---------------------------------------------------------------------------
// act = silu(gate) * up, in place over gate buffer
// ---------------------------------------------------------------------------
__global__ __launch_bounds__(256)
void silu_mul_kernel(bf16* __restrict__ G, const bf16* __restrict__ U,
                     long long n)
{
  long long i = (long long)blockIdx.x * 256 + threadIdx.x;
  if (i >= n) return;
  float g = bf2f(G[i]);
  float u = bf2f(U[i]);
  float s = g / (1.0f + __expf(-g));
  G[i] = f2bf(s * u);
}

// ---------------------------------------------------------------------------
extern "C" void kernel_launch(void* const* d_in, const int* in_sizes, int n_in,
                              void* d_out, int out_size, void* d_ws,
                              size_t ws_size, hipStream_t stream)
{
  (void)in_sizes; (void)n_in; (void)out_size; (void)ws_size;
  const float* x        = (const float*)d_in[0];
  const float* norm_w   = (const float*)d_in[1];
  const float* attn_w   = (const float*)d_in[2];
  const float* ffn_up   = (const float*)d_in[3];
  const float* ffn_gate = (const float*)d_in[4];
  const float* ffn_down = (const float*)d_in[5];
  const unsigned char* masks[4] = {
      (const unsigned char*)d_in[6], (const unsigned char*)d_in[7],
      (const unsigned char*)d_in[8], (const unsigned char*)d_in[9]};

  char* ws = (char*)d_ws;
  size_t off = 0;
  auto alloc = [&](size_t bytes) {
    char* p = ws + off;
    off += (bytes + 255) & ~(size_t)255;
    return p;
  };
  const size_t TOK = (size_t)BATCH * SEQ;   // 4096 tokens

  float* hcur = (float*)alloc(TOK * DIM * 4);        // residual stream f32
  bf16* hn    = (bf16*)alloc(TOK * DIM * 2);
  bf16* qb    = (bf16*)alloc(TOK * DIM * 2);         // Q (B,S,D)
  bf16* kb    = (bf16*)alloc(TOK * DIM * 2);         // K (B,S,D)
  bf16* vt    = (bf16*)alloc(TOK * DIM * 2);         // V^T (B,D,S)
  bf16* aob   = (bf16*)alloc(TOK * DIM * 2);         // attention out (B,S,D)
  bf16* gbuf  = (bf16*)alloc(TOK * DFFN * 2);        // gate / act
  bf16* ubuf  = (bf16*)alloc(TOK * DFFN * 2);        // up
  bf16* wt_attn = (bf16*)alloc((size_t)16 * DIM * DIM * 2);  // per-layer W^T
  bf16* wt_gate = (bf16*)alloc((size_t)DIM * DFFN * 2);
  bf16* wt_up   = (bf16*)alloc((size_t)DIM * DFFN * 2);
  bf16* wt_down = (bf16*)alloc((size_t)DFFN * DIM * 2);

  hipMemcpyAsync(hcur, x, TOK * DIM * 4, hipMemcpyDeviceToDevice, stream);

  const int TB = 256;
  const int gwD  = (int)(TOK / 256) * (DIM / 64);    // 128 blocks
  const int gwF  = (int)(TOK / 256) * (DFFN / 64);   // 512 blocks
  const int gwAt = (BATCH * NHEAD * (SEQ / 16)) / 8; // 256 blocks
  const int gcDD = (DIM / 32) * (DIM / 32);          // 256 blocks
  const int gcDF = (DIM / 32) * (DFFN / 32);         // 1024 blocks

  for (int l = 0; l < NLAYER; ++l) {
    // --- convert this layer's weights to transposed bf16 ---
    for (int i = 0; i < 4; ++i)
      for (int w = 0; w < 4; ++w) {
        const float* W = attn_w + (((size_t)l * 4 + i) * 4 + w) * DIM * DIM;
        bf16* Wt = wt_attn + (size_t)(i * 4 + w) * DIM * DIM;
        convw_kernel<<<gcDD, TB, 0, stream>>>(W, Wt, DIM, DIM);
      }
    convw_kernel<<<gcDF, TB, 0, stream>>>(
        ffn_gate + (size_t)l * DIM * DFFN, wt_gate, DIM, DFFN);
    convw_kernel<<<gcDF, TB, 0, stream>>>(
        ffn_up + (size_t)l * DIM * DFFN, wt_up, DIM, DFFN);
    convw_kernel<<<gcDF, TB, 0, stream>>>(
        ffn_down + (size_t)l * DFFN * DIM, wt_down, DFFN, DIM);

    // --- 4 masked attention sub-blocks ---
    for (int i = 0; i < 4; ++i) {
      const float* nw = norm_w + ((size_t)l * 5 + i) * DIM;
      rmsnorm_kernel<<<(int)TOK, TB, 0, stream>>>(hcur, nw, hn);
      const bf16* wq = wt_attn + (size_t)(i * 4 + 0) * DIM * DIM;
      const bf16* wk = wt_attn + (size_t)(i * 4 + 1) * DIM * DIM;
      const bf16* wv = wt_attn + (size_t)(i * 4 + 2) * DIM * DIM;
      const bf16* wo = wt_attn + (size_t)(i * 4 + 3) * DIM * DIM;
      gemm_kernel<<<gwD, TB, 0, stream>>>(hn, wq, (int)TOK, DIM, DIM, qb, 0, DIM);
      gemm_kernel<<<gwD, TB, 0, stream>>>(hn, wk, (int)TOK, DIM, DIM, kb, 0, DIM);
      gemm_kernel<<<gwD, TB, 0, stream>>>(hn, wv, (int)TOK, DIM, DIM, vt, 1, DIM);
      attn_kernel<<<gwAt, TB, 0, stream>>>(qb, kb, vt, masks[i], aob);
      gemm_kernel<<<gwD, TB, 0, stream>>>(aob, wo, (int)TOK, DIM, DIM, hcur, 2, DIM);
    }

    // --- FFN ---
    const float* nw4 = norm_w + ((size_t)l * 5 + 4) * DIM;
    rmsnorm_kernel<<<(int)TOK, TB, 0, stream>>>(hcur, nw4, hn);
    gemm_kernel<<<gwF, TB, 0, stream>>>(hn, wt_gate, (int)TOK, DFFN, DIM, gbuf, 0, DFFN);
    gemm_kernel<<<gwF, TB, 0, stream>>>(hn, wt_up,   (int)TOK, DFFN, DIM, ubuf, 0, DFFN);
    long long nel = (long long)TOK * DFFN;
    silu_mul_kernel<<<(int)(nel / TB), TB, 0, stream>>>(gbuf, ubuf, nel);
    gemm_kernel<<<gwD, TB, 0, stream>>>(gbuf, wt_down, (int)TOK, DIM, DFFN, hcur, 2, DIM);
  }

  hipMemcpyAsync(d_out, hcur, TOK * DIM * 4, hipMemcpyDeviceToDevice, stream);
}
